// SharedVocabCopyGenerator_86500641341806
// MI455X (gfx1250) — matrix operand order repaired and checked by reading, not compiled, not run on hardware
//
#include <hip/hip_runtime.h>
#include <hip/hip_bf16.h>
#include <math.h>

// Problem dims (fixed by reference setup_inputs)
constexpr int N = 1024;    // rows (tlen*batch)
constexpr int D = 1024;    // hidden size (K of GEMM)
constexpr int S = 64;      // source length
constexpr int B = 64;      // batch
constexpr int V = 32000;   // vocab (N of GEMM)
constexpr int PAD = 1;

typedef __attribute__((ext_vector_type(16))) __bf16 v16bf;
typedef __attribute__((ext_vector_type(8)))  float  v8f;

// ---------------------------------------------------------------------------
// K1: p_copy[r] = sigmoid(hidden[r,:] . w_copy + b)
// ---------------------------------------------------------------------------
__global__ void __launch_bounds__(128)
pcopy_kernel(const float* __restrict__ hidden, const float* __restrict__ wc,
             const float* __restrict__ bc, float* __restrict__ p) {
  const int row = blockIdx.x;
  const int t = threadIdx.x;
  float s = 0.f;
  #pragma unroll
  for (int j = 0; j < D / 128; ++j)
    s += hidden[(size_t)row * D + t + 128 * j] * wc[t + 128 * j];
  #pragma unroll
  for (int off = 16; off; off >>= 1) s += __shfl_down(s, off, 32);
  __shared__ float red[4];
  if ((t & 31) == 0) red[t >> 5] = s;
  __syncthreads();
  if (t == 0) {
    float tot = red[0] + red[1] + red[2] + red[3] + bc[0];
    p[row] = 1.0f / (1.0f + __expf(-tot));
  }
}

// ---------------------------------------------------------------------------
// K2: logits = hidden @ W^T  (raw f32 logits written to d_out)
// bf16 WMMA, 128x256 block tile, K-steps of 32 staged through LDS as bf16.
// 8 waves in a 2x4 grid; each wave owns a 64x64 strip = 4x4 16x16 tiles
// (16 WMMAs per K-step against 16 b128 fragment loads).
// ---------------------------------------------------------------------------
constexpr int BLK_M = 128;
constexpr int BLK_N = 256;
constexpr int TILE_K = 32;
constexpr int LDSROW = 40;   // 32 bf16 data + 8 pad (80B row -> conflict-free b128)
constexpr int TOT_ROWS = BLK_M + BLK_N;  // A rows [0,128), W rows [128,384)

union FragCvt {    // 32 bytes: two 16B chunks <-> one 16-elem bf16 fragment
  uint4 u[2];
  v16bf v;
};

// load 16 f32 from global, convert to bf16, store 32B into LDS
__device__ __forceinline__ void stage16(const float* __restrict__ g,
                                        __bf16* __restrict__ l) {
  float f[16];
  *(float4*)&f[0]  = *(const float4*)(g);
  *(float4*)&f[4]  = *(const float4*)(g + 4);
  *(float4*)&f[8]  = *(const float4*)(g + 8);
  *(float4*)&f[12] = *(const float4*)(g + 12);
  union { __bf16 h[16]; uint4 u[2]; } cv;
  #pragma unroll
  for (int i = 0; i < 16; ++i) cv.h[i] = (__bf16)f[i];
  *(uint4*)(l)     = cv.u[0];
  *(uint4*)(l + 8) = cv.u[1];
}

__global__ void __launch_bounds__(256)
gemm_logits_kernel(const float* __restrict__ hidden, const float* __restrict__ W,
                   float* __restrict__ out) {
  __shared__ __bf16 Ts[TOT_ROWS * LDSROW];

  const int tid  = threadIdx.x;
  const int wave = tid >> 5;
  const int lane = tid & 31;
  const int wm   = wave >> 2;        // 0..1  -> 64-row strip
  const int wn   = wave & 3;         // 0..3  -> 64-col strip
  const int ln   = lane & 15;
  const int lhalf = lane >> 4;       // 0/1

  const int m0 = blockIdx.y * BLK_M;
  const int n0 = blockIdx.x * BLK_N;

  v8f acc[4][4] = {};                // 4 M-subtiles x 4 N-subtiles (16x16 each)

  // cooperative staging: 384 rows x 2 k-halves = 768 tasks, 3 per thread
  const int srow = tid >> 1;         // 0..127
  const int skh  = (tid & 1) * 16;   // k offset 0 or 16
  const float* ga  = hidden + (size_t)(m0 + srow) * D + skh;        // A row
  const float* gb0 = W + (size_t)(n0 + srow) * D + skh;             // W rows 0..127
  const float* gb1 = W + (size_t)(n0 + 128 + srow) * D + skh;       // W rows 128..255
  __bf16* la  = &Ts[srow * LDSROW + skh];
  __bf16* lb0 = &Ts[(BLK_M + srow) * LDSROW + skh];
  __bf16* lb1 = &Ts[(BLK_M + 128 + srow) * LDSROW + skh];

  for (int k0 = 0; k0 < D; k0 += TILE_K) {
    stage16(ga + k0, la);
    stage16(gb0 + k0, lb0);
    stage16(gb1 + k0, lb1);
    if (k0 + TILE_K < D) {           // prefetch next K-panel (global_prefetch_b8)
      __builtin_prefetch(ga + k0 + TILE_K, 0, 1);
      __builtin_prefetch(gb0 + k0 + TILE_K, 0, 1);
      __builtin_prefetch(gb1 + k0 + TILE_K, 0, 1);
    }
    __syncthreads();

    // Build fragments.
    // A layout (ISA 16-bit A 16x32): lane ln row M=ln; elems 0-7 at k=8*lhalf+e,
    // elems 8-15 at k=16+8*lhalf+(e-8)  -> two 16B LDS reads.
    FragCvt fa[4];
    #pragma unroll
    for (int ms = 0; ms < 4; ++ms) {
      const __bf16* arow = &Ts[(wm * 64 + ms * 16 + ln) * LDSROW];
      fa[ms].u[0] = *(const uint4*)(arow + 8 * lhalf);
      fa[ms].u[1] = *(const uint4*)(arow + 16 + 8 * lhalf);
    }
    // B layout: lane ln col N=ln; elems 0-7 at k=8*lhalf+e,
    // elems 8-15 at k=16+8*lhalf+(e-8) -> two 16B LDS reads (same as A).
    FragCvt fb[4];
    #pragma unroll
    for (int ns = 0; ns < 4; ++ns) {
      const __bf16* brow = &Ts[(BLK_M + wn * 64 + ns * 16 + ln) * LDSROW];
      fb[ns].u[0] = *(const uint4*)(brow + 8 * lhalf);
      fb[ns].u[1] = *(const uint4*)(brow + 16 + 8 * lhalf);
    }
    #pragma unroll
    for (int ms = 0; ms < 4; ++ms)
      #pragma unroll
      for (int ns = 0; ns < 4; ++ns)
        acc[ms][ns] = __builtin_amdgcn_wmma_f32_16x16x32_bf16(
            false, fa[ms].v, false, fb[ns].v, (short)0, acc[ms][ns],
            false, false);
    __syncthreads();
  }

  // store raw logits; C layout: VGPR i -> row = i + 8*lhalf, col = ln
  #pragma unroll
  for (int ms = 0; ms < 4; ++ms) {
    const int rbase = m0 + wm * 64 + ms * 16 + lhalf * 8;
    #pragma unroll
    for (int ns = 0; ns < 4; ++ns) {
      const int col = n0 + wn * 64 + ns * 16 + ln;
      #pragma unroll
      for (int i = 0; i < 8; ++i)
        out[(size_t)(rbase + i) * V + col] = acc[ms][ns][i];
    }
  }
}

// ---------------------------------------------------------------------------
// K3: per-row online (max, sum-exp) over logits, excluding PAD column
// ---------------------------------------------------------------------------
__device__ inline void merge_ms(float& m, float& s, float m2, float s2) {
  float M = fmaxf(m, m2);
  if (M == -INFINITY) { s = 0.f; m = M; return; }
  s = s * __expf(m - M) + s2 * __expf(m2 - M);
  m = M;
}

__global__ void __launch_bounds__(256)
rowstats_kernel(const float* __restrict__ logits, float* __restrict__ rmax,
                float* __restrict__ rinv) {
  const int row = blockIdx.x;
  const int t = threadIdx.x;
  const float* L = logits + (size_t)row * V;
  float m = -INFINITY, s = 0.f;
  for (int c = t; c < V; c += 256) {
    if (c == PAD) continue;
    float x = L[c];
    if (x > m) { s = s * __expf(m - x) + 1.0f; m = x; }
    else       { s += __expf(x - m); }
  }
  #pragma unroll
  for (int off = 16; off; off >>= 1) {
    float m2 = __shfl_down(m, off, 32);
    float s2 = __shfl_down(s, off, 32);
    merge_ms(m, s, m2, s2);
  }
  __shared__ float sm[8], ss[8];
  if ((t & 31) == 0) { sm[t >> 5] = m; ss[t >> 5] = s; }
  __syncthreads();
  if (t == 0) {
    #pragma unroll
    for (int w = 1; w < 8; ++w) merge_ms(m, s, sm[w], ss[w]);
    rmax[row] = m;
    rinv[row] = 1.0f / s;
  }
}

// ---------------------------------------------------------------------------
// K4: in-place finalize: out = (1-p)*(softmax + 1e-20), PAD col prob = 0
// ---------------------------------------------------------------------------
__global__ void __launch_bounds__(256)
finalize_kernel(float* __restrict__ out, const float* __restrict__ p,
                const float* __restrict__ rmax, const float* __restrict__ rinv) {
  const size_t g = (size_t)blockIdx.x * 256 + threadIdx.x;  // float4 index
  const int row = (int)(g / (V / 4));
  const int c0  = (int)(g % (V / 4)) * 4;
  const float m = rmax[row];
  const float is = rinv[row];
  const float gate = 1.0f - p[row];
  float4* ptr = (float4*)(out + (size_t)row * V + c0);
  float4 v = *ptr;
  float x[4] = {v.x, v.y, v.z, v.w};
  #pragma unroll
  for (int i = 0; i < 4; ++i) {
    float prob = (c0 + i == PAD) ? 0.f : __expf(x[i] - m) * is;
    x[i] = gate * (prob + 1e-20f);
  }
  *ptr = make_float4(x[0], x[1], x[2], x[3]);
}

// ---------------------------------------------------------------------------
// K5: copy-attention scatter: out[r, src[s, r%B]] += p[r] * attn[r, s]
// ---------------------------------------------------------------------------
__global__ void __launch_bounds__(256)
scatter_kernel(const float* __restrict__ attn, const int* __restrict__ src,
               const float* __restrict__ p, float* __restrict__ out) {
  const int g = blockIdx.x * 256 + threadIdx.x;   // 0 .. N*S-1
  const int r = g >> 6;          // row
  const int s = g & 63;          // source position
  const int b = r & (B - 1);     // batch index
  const int v = src[s * B + b];  // vocab id
  const float w = p[r] * attn[(size_t)r * S + s];
  atomicAdd(out + (size_t)r * V + v, w);
}

// ---------------------------------------------------------------------------
extern "C" void kernel_launch(void* const* d_in, const int* in_sizes, int n_in,
                              void* d_out, int out_size, void* d_ws, size_t ws_size,
                              hipStream_t stream) {
  const float* hidden = (const float*)d_in[0];   // [N, D]
  const float* attn   = (const float*)d_in[1];   // [N, S]
  const int*   src    = (const int*)d_in[2];     // [S, B, 1]
  const float* W      = (const float*)d_in[3];   // [V, D]
  const float* wcopy  = (const float*)d_in[4];   // [D, 1]
  const float* bcopy  = (const float*)d_in[5];   // [1]
  float* out = (float*)d_out;                    // [N, V]

  float* ws   = (float*)d_ws;
  float* p    = ws;            // N floats
  float* rmax = ws + N;        // N floats
  float* rinv = ws + 2 * N;    // N floats

  pcopy_kernel<<<N, 128, 0, stream>>>(hidden, wcopy, bcopy, p);

  dim3 gg(V / BLK_N, N / BLK_M);   // 125 x 8
  gemm_logits_kernel<<<gg, 256, 0, stream>>>(hidden, W, out);

  rowstats_kernel<<<N, 256, 0, stream>>>(out, rmax, rinv);

  finalize_kernel<<<(N * (V / 4)) / 256, 256, 0, stream>>>(out, p, rmax, rinv);

  scatter_kernel<<<(N * S) / 256, 256, 0, stream>>>(attn, src, p, out);
}
